// LocalFoldingGraph_42073499631911
// MI455X (gfx1250) — matrix operand order repaired and checked by reading, not compiled, not run on hardware
//
#include <hip/hip_runtime.h>
#include <hip/hip_bf16.h>

typedef _Float16 v16h __attribute__((ext_vector_type(16)));
typedef _Float16 v8h  __attribute__((ext_vector_type(8)));
typedef float    v8f  __attribute__((ext_vector_type(8)));

#define BB 64
#define JJ 21
#define NN 2048
#define CC 256
#define NS 64
#define QP (BB*JJ)          // 1344 (b,j) pairs
#define PP (QP*NS)          // 86016 pixels
#define KPAD 544            // padded K for conv layers (521 / 518 -> 544)
#define LEAK 0.1f

// GEMM macro-tile config: block = 8 waves, 32(M) x 512(N) per block,
// each wave = 2(M) x 4(N) subtiles of 16x16.
#define GMT   32
#define GNTW  64
#define GWAVES 8
#define GNT   (GWAVES*GNTW)   // 512

// ---------------------------------------------------------------------------
// KNN: one block per (b,j); exact iterative argmin-64 over 2048 distances.
// Order of the 64 indices is irrelevant downstream (pointwise + max over s).
// ---------------------------------------------------------------------------
__global__ void __launch_bounds__(256) knn_kernel(
    const float* __restrict__ joints, const float* __restrict__ xyz,
    int* __restrict__ knn)
{
  int q = blockIdx.x; int b = q / JJ; int j = q % JJ;
  __shared__ float dist[NN];
  __shared__ float sval[256];
  __shared__ int   sidx[256];
  float jx = joints[b*3*JJ + 0*JJ + j];
  float jy = joints[b*3*JJ + 1*JJ + j];
  float jz = joints[b*3*JJ + 2*JJ + j];
  const float* xb = xyz + (size_t)b*3*NN;
  for (int n = threadIdx.x; n < NN; n += 256) {
    float dx = xb[n] - jx, dy = xb[NN+n] - jy, dz = xb[2*NN+n] - jz;
    dist[n] = dx*dx + dy*dy + dz*dz;
  }
  __syncthreads();
  for (int r = 0; r < NS; ++r) {
    float bv = 3.4e38f; int bi = 0x7fffffff;
    for (int n = threadIdx.x; n < NN; n += 256) {
      float v = dist[n];
      if (v < bv || (v == bv && n < bi)) { bv = v; bi = n; }
    }
    sval[threadIdx.x] = bv; sidx[threadIdx.x] = bi;
    __syncthreads();
    for (int off = 128; off > 0; off >>= 1) {
      if (threadIdx.x < off) {
        float ov = sval[threadIdx.x + off]; int oi = sidx[threadIdx.x + off];
        if (ov < sval[threadIdx.x] ||
            (ov == sval[threadIdx.x] && oi < sidx[threadIdx.x])) {
          sval[threadIdx.x] = ov; sidx[threadIdx.x] = oi;
        }
      }
      __syncthreads();
    }
    if (threadIdx.x == 0) { knn[q*NS + r] = sidx[0]; dist[sidx[0]] = 3.4e38f; }
    __syncthreads();
  }
}

// ---------------------------------------------------------------------------
// f32 -> f16 plain convert (gw / m2w weights; K already multiple of 32)
// ---------------------------------------------------------------------------
__global__ void f32_to_f16_kernel(const float* __restrict__ src,
                                  _Float16* __restrict__ dst, int n)
{
  int i = blockIdx.x * blockDim.x + threadIdx.x;
  if (i < n) dst[i] = (_Float16)src[i];
}

// ---------------------------------------------------------------------------
// Pack conv weights into permuted, KPAD-padded f16 layout matching the X
// channel order (conv output channels first so GEMM writes land at offset 0).
//   layer0 X: [grouped 0..255][dir 256..258][time 259..261][lat 262..517][je 518..520][pad]
//   layer1/2 X: [prev 0..255][time 256..258][lat 259..514][je 515..517][pad]
// reference order: [time 0..2][lat 3..258][je 259..261][grouped/prev 262..517][dir 518..520]
// ---------------------------------------------------------------------------
__global__ void pack_conv_w_kernel(const float* __restrict__ cw,
                                   _Float16* __restrict__ Wp,
                                   int layer, int Kref)
{
  int o = blockIdx.x;  // 0..255
  for (int x = threadIdx.x; x < KPAD; x += blockDim.x) {
    int r = -1;
    if (layer == 0) {
      if      (x < 256) r = 262 + x;          // grouped
      else if (x < 259) r = 518 + (x - 256);  // direction
      else if (x < 262) r = x - 259;          // time
      else if (x < 518) r = 3 + (x - 262);    // lat
      else if (x < 521) r = 259 + (x - 518);  // joint emb
    } else {
      if      (x < 256) r = 262 + x;          // prev conv out
      else if (x < 259) r = x - 256;          // time
      else if (x < 515) r = 3 + (x - 259);    // lat
      else if (x < 518) r = 259 + (x - 515);  // joint emb
    }
    Wp[(size_t)o*KPAD + x] = (r >= 0) ? (_Float16)cw[(size_t)o*Kref + r]
                                      : (_Float16)0.f;
  }
}

// ---------------------------------------------------------------------------
// Graph adjacency: tmpT[(b*21+k)][d] = sum_j latents[b,d,j] * A[d,j,k]  (f16)
// grid = 1344 (b,k), block = 256 (d)
// ---------------------------------------------------------------------------
__global__ void __launch_bounds__(256) graph_kernel(
    const float* __restrict__ lat, const float* __restrict__ A,
    _Float16* __restrict__ tmpT)
{
  int bk = blockIdx.x; int b = bk / JJ; int k = bk % JJ;
  int d = threadIdx.x;
  const float* lr = lat + ((size_t)b*CC + d)*JJ;
  const float* ar = A + (size_t)d*JJ*JJ + k;
  float acc = 0.f;
#pragma unroll
  for (int j = 0; j < JJ; ++j) acc += lr[j] * ar[j*JJ];
  tmpT[(size_t)bk*CC + d] = (_Float16)acc;
}

// ---------------------------------------------------------------------------
// Build the conv input matrix X[p][KPAD] (pixel-major, f16).
// grid = 86016 (p), block = 256
// ---------------------------------------------------------------------------
__global__ void __launch_bounds__(256) build_x_kernel(
    int layer,
    const float* __restrict__ points, const float* __restrict__ xyz,
    const float* __restrict__ joints, const int* __restrict__ knn,
    const _Float16* __restrict__ Y, const _Float16* __restrict__ latbuf,
    const float* __restrict__ beta, const float* __restrict__ joint_idx,
    _Float16* __restrict__ X)
{
  int p = blockIdx.x;
  int s = p & (NS-1); int q = p >> 6; int b = q / JJ; int j = q % JJ;
  _Float16* xr = X + (size_t)p * KPAD;
  const _Float16* latq = latbuf + (size_t)q * CC;
  float bt = beta[b];
  float ji = joint_idx[j];
  if (layer == 0) {
    int idx = knn[q*NS + s];
    for (int c = threadIdx.x; c < CC; c += 256)
      xr[c] = (_Float16)points[((size_t)b*CC + c)*NN + idx];
    for (int c = threadIdx.x; c < CC; c += 256)
      xr[262 + c] = latq[c];
    if (threadIdx.x < 3) {
      int c = threadIdx.x;
      float nv = xyz[((size_t)b*3 + c)*NN + idx];
      float jt = joints[b*3*JJ + c*JJ + j];
      xr[256 + c] = (_Float16)(nv - jt);                               // direction
      float te = (c == 0) ? bt : ((c == 1) ? sinf(bt) : cosf(bt));
      xr[259 + c] = (_Float16)te;                                      // time emb
      float je = (c == 0) ? ji : ((c == 1) ? sinf(ji) : cosf(ji));
      xr[518 + c] = (_Float16)je;                                      // joint emb
    }
    if (threadIdx.x < KPAD - 521) xr[521 + threadIdx.x] = (_Float16)0.f;
  } else {
    const _Float16* yr = Y + (size_t)p * CC;
    for (int c = threadIdx.x; c < CC; c += 256) xr[c] = yr[c];
    for (int c = threadIdx.x; c < CC; c += 256) xr[259 + c] = latq[c];
    if (threadIdx.x < 3) {
      int c = threadIdx.x;
      float te = (c == 0) ? bt : ((c == 1) ? sinf(bt) : cosf(bt));
      xr[256 + c] = (_Float16)te;
      float je = (c == 0) ? ji : ((c == 1) ? sinf(ji) : cosf(ji));
      xr[515 + c] = (_Float16)je;
    }
    if (threadIdx.x < KPAD - 518) xr[518 + threadIdx.x] = (_Float16)0.f;
  }
}

// ---------------------------------------------------------------------------
// WMMA f16 GEMM:  Out[m,p] = act( sum_k W[m][k] * X[p][k] + bias[m] )
// Block = 8 waves, macro-tile 32(M) x 512(N).
// W panel (32 x Kpad f16, <= 34.8 KB) is staged in LDS once and shared by all
// 8 waves (A fragments via ds_load_b128). Each wave computes 2x4 subtiles of
// 16x16, reusing each A frag 4x and each B frag 2x: 8 WMMAs per 32-K step on
// 4 KB of global traffic (~32 FLOP/B from L2).
// Fragment addressing (same contiguous K order for A and B):
//   A lane l (row m = l&15): halves 0..15 = K kk+(l>=16)*16 .. +15 (contiguous)
//   B lane l (col p = l&15): halves 0..15 = K kk+(l>=16)*16 .. +15 (contiguous)
// ---------------------------------------------------------------------------
__global__ void __launch_bounds__(256) wmma_gemm_kernel(
    const _Float16* __restrict__ W, const _Float16* __restrict__ X,
    const float* __restrict__ bias, int Kpad, int ldX, int P,
    _Float16* __restrict__ outF16, int ldOut,
    float* __restrict__ outF32, int M)
{
  __shared__ __align__(16) _Float16 lw[GMT * KPAD];   // 34,816 B max
  int tid = threadIdx.x;

  // ---- stage W panel (contiguous GMT*Kpad halves) into LDS ----
  {
    const _Float16* wpanel = W + (size_t)(blockIdx.y * GMT) * Kpad;
    int total = GMT * Kpad;                  // multiple of 8
    for (int i = tid * 8; i < total; i += 256 * 8)
      *(v8h*)(lw + i) = *(const v8h*)(wpanel + i);
  }
  __syncthreads();

  int wave = tid >> 5;
  int lane = tid & 31;
  int hi   = lane >> 4;        // 0 or 1
  int lm   = lane & 15;
  int ptile = blockIdx.x * GNT + wave * GNTW;
  if (ptile >= P) return;

  const _Float16* la  = lw + (size_t)lm * Kpad + hi * 16;
  const _Float16* xr0 = X + (size_t)(ptile + lm) * ldX + hi * 16;

  v8f acc[2][4] = {};
  for (int kk = 0; kk < Kpad; kk += 32) {
    __builtin_prefetch((const void*)(xr0 + kk + 128), 0, 0);
    v16h a[2];
#pragma unroll
    for (int mi = 0; mi < 2; ++mi) {
      const _Float16* ap = la + (size_t)(mi * 16) * Kpad + kk;
      v8h alo = *(const v8h*)(ap);          // K = kk+hi*16 .. +7   (LDS)
      v8h ahi = *(const v8h*)(ap + 8);      // K = kk+hi*16+8 .. +15
      a[mi] = __builtin_shufflevector(alo, ahi, 0,1,2,3,4,5,6,7,
                                                8,9,10,11,12,13,14,15);
    }
    v16h bf[4];
#pragma unroll
    for (int ni = 0; ni < 4; ++ni)
      bf[ni] = *(const v16h*)(xr0 + (size_t)(ni * 16) * ldX + kk);
#pragma unroll
    for (int mi = 0; mi < 2; ++mi)
#pragma unroll
      for (int ni = 0; ni < 4; ++ni)
        acc[mi][ni] = __builtin_amdgcn_wmma_f32_16x16x32_f16(
            false, a[mi], false, bf[ni], (short)0, acc[mi][ni], false, false);
  }

#pragma unroll
  for (int mi = 0; mi < 2; ++mi) {
    int mbase = blockIdx.y * GMT + mi * 16 + hi * 8;  // C/D: vgpr v -> mbase+v
#pragma unroll
    for (int ni = 0; ni < 4; ++ni) {
      int p = ptile + ni * 16 + lm;
      if (outF16) {
        v8h ov;
#pragma unroll
        for (int v = 0; v < 8; ++v) {
          float val = acc[mi][ni][v] + (bias ? bias[mbase + v] : 0.f);
          val = (val >= 0.f) ? val : LEAK * val;
          ov[v] = (_Float16)val;
        }
        *(v8h*)(outF16 + (size_t)p * ldOut + mbase) = ov;   // 16B aligned
      } else {
        int b = p / JJ, j = p % JJ;
        float* orow = outF32 + (size_t)b * M * JJ + j;
#pragma unroll
        for (int v = 0; v < 8; ++v) {
          float val = acc[mi][ni][v];
          val = (val >= 0.f) ? val : LEAK * val;
          orow[(size_t)(mbase + v) * JJ] = val;
        }
      }
    }
  }
}

// ---------------------------------------------------------------------------
// Maxpool over s: Y[p][c] -> lat32[b][c][j] (f32) and featT[q][c] (f16)
// ---------------------------------------------------------------------------
__global__ void __launch_bounds__(256) maxpool_kernel(
    const _Float16* __restrict__ Y, float* __restrict__ lat32,
    _Float16* __restrict__ featT)
{
  int q = blockIdx.x; int b = q / JJ; int j = q % JJ; int c = threadIdx.x;
  const _Float16* yb = Y + (size_t)q * NS * CC + c;
  float m = -3.4e38f;
  for (int s = 0; s < NS; ++s) {
    float v = (float)yb[(size_t)s * CC];
    m = (v > m) ? v : m;
  }
  lat32[((size_t)b*CC + c)*JJ + j] = m;
  featT[(size_t)q*CC + c] = (_Float16)m;
}

// ---------------------------------------------------------------------------
// Regress head: joint_res = regw @ feats + regb + joints  (tiny)
// ---------------------------------------------------------------------------
__global__ void regress_kernel(const float* __restrict__ feats,
                               const float* __restrict__ regw,
                               const float* __restrict__ regb,
                               const float* __restrict__ joints,
                               float* __restrict__ out)
{
  int t = blockIdx.x * blockDim.x + threadIdx.x;
  if (t >= BB*3*JJ) return;
  int j = t % JJ; int r = (t / JJ) % 3; int b = t / (3*JJ);
  const float* f = feats + (size_t)b*512*JJ + j;
  const float* w = regw + (size_t)r*512;
  float acc = regb[r] + joints[(size_t)b*3*JJ + r*JJ + j];
  for (int c = 0; c < 512; ++c) acc += w[c] * f[(size_t)c*JJ];
  out[(size_t)b*3*JJ + r*JJ + j] = acc;
}

// ---------------------------------------------------------------------------
extern "C" void kernel_launch(void* const* d_in, const int* in_sizes, int n_in,
                              void* d_out, int out_size, void* d_ws, size_t ws_size,
                              hipStream_t stream)
{
  const float* joints    = (const float*)d_in[0];
  const float* xyz       = (const float*)d_in[1];
  const float* points    = (const float*)d_in[2];
  const float* beta      = (const float*)d_in[3];
  const float* joint_idx = (const float*)d_in[4];
  const float* latents0  = (const float*)d_in[5];
  const float* A[3]  = {(const float*)d_in[6],  (const float*)d_in[7],  (const float*)d_in[8]};
  const float* gw[3] = {(const float*)d_in[9],  (const float*)d_in[11], (const float*)d_in[13]};
  const float* gb[3] = {(const float*)d_in[10], (const float*)d_in[12], (const float*)d_in[14]};
  const float* cw[3] = {(const float*)d_in[15], (const float*)d_in[17], (const float*)d_in[19]};
  const float* cb[3] = {(const float*)d_in[16], (const float*)d_in[18], (const float*)d_in[20]};
  const float* m2w0 = (const float*)d_in[21];
  const float* m2w1 = (const float*)d_in[22];
  const float* regw = (const float*)d_in[23];
  const float* regb = (const float*)d_in[24];
  float* out = (float*)d_out;   // [joint_res 4032][feats 688128]

  char* wsp = (char*)d_ws;
  auto alloc = [&](size_t bytes) -> char* {
    char* p = wsp; wsp += (bytes + 255) & ~(size_t)255; return p;
  };
  int*      knn    = (int*)     alloc((size_t)QP*NS*4);
  _Float16* X      = (_Float16*)alloc((size_t)PP*KPAD*2);   // ~93.6 MB
  _Float16* Y      = (_Float16*)alloc((size_t)PP*CC*2);     // ~44 MB
  _Float16* tmpT   = (_Float16*)alloc((size_t)QP*CC*2);
  _Float16* latbuf = (_Float16*)alloc((size_t)QP*CC*2);
  float*    lat32  = (float*)   alloc((size_t)BB*CC*JJ*4);
  _Float16* featT  = (_Float16*)alloc((size_t)QP*CC*2);
  _Float16* H1     = (_Float16*)alloc((size_t)QP*512*2);
  _Float16* WpC[3]; for (int i = 0; i < 3; ++i) WpC[i] = (_Float16*)alloc((size_t)CC*KPAD*2);
  _Float16* gwF[3]; for (int i = 0; i < 3; ++i) gwF[i] = (_Float16*)alloc((size_t)CC*CC*2);
  _Float16* m2F0 = (_Float16*)alloc((size_t)512*CC*2);
  _Float16* m2F1 = (_Float16*)alloc((size_t)512*512*2);

  // -- weight preparation --------------------------------------------------
  for (int L = 0; L < 3; ++L) {
    pack_conv_w_kernel<<<CC, 256, 0, stream>>>(cw[L], WpC[L], L, (L == 0) ? 521 : 518);
    f32_to_f16_kernel<<<(CC*CC + 255)/256, 256, 0, stream>>>(gw[L], gwF[L], CC*CC);
  }
  f32_to_f16_kernel<<<(512*CC  + 255)/256, 256, 0, stream>>>(m2w0, m2F0, 512*CC);
  f32_to_f16_kernel<<<(512*512 + 255)/256, 256, 0, stream>>>(m2w1, m2F1, 512*512);

  // -- KNN -----------------------------------------------------------------
  knn_kernel<<<QP, 256, 0, stream>>>(joints, xyz, knn);

  // -- 3 graph/conv layers -------------------------------------------------
  for (int L = 0; L < 3; ++L) {
    const float* latSrc = (L == 0) ? latents0 : lat32;
    graph_kernel<<<QP, 256, 0, stream>>>(latSrc, A[L], tmpT);
    // lat = leaky(gw @ tmp + gb): M=256, K=256, P=1344
    wmma_gemm_kernel<<<dim3((QP + GNT - 1)/GNT, CC/GMT), 256, 0, stream>>>(
        gwF[L], tmpT, gb[L], CC, CC, QP, latbuf, CC, nullptr, CC);
    build_x_kernel<<<PP, 256, 0, stream>>>(
        L, points, xyz, joints, knn, Y, latbuf, beta, joint_idx, X);
    // conv: M=256, Kpad=544, P=86016
    wmma_gemm_kernel<<<dim3(PP/GNT, CC/GMT), 256, 0, stream>>>(
        WpC[L], X, cb[L], KPAD, KPAD, PP, Y, CC, nullptr, CC);
    maxpool_kernel<<<QP, 256, 0, stream>>>(Y, lat32, featT);
  }

  // -- mlp2 ----------------------------------------------------------------
  // H1 = leaky(m2w0 @ feats): M=512, K=256, P=1344
  wmma_gemm_kernel<<<dim3((QP + GNT - 1)/GNT, 512/GMT), 256, 0, stream>>>(
      m2F0, featT, nullptr, CC, CC, QP, H1, 512, nullptr, 512);
  // feats_out = leaky(m2w1 @ H1): M=512, K=512, f32 out -> d_out[4032:]
  wmma_gemm_kernel<<<dim3((QP + GNT - 1)/GNT, 512/GMT), 256, 0, stream>>>(
      m2F1, H1, nullptr, 512, 512, QP, nullptr, 0, out + BB*3*JJ, 512);

  // -- regression head -----------------------------------------------------
  regress_kernel<<<(BB*3*JJ + 63)/64, 64, 0, stream>>>(
      out + BB*3*JJ, regw, regb, joints, out);
}
